// AdaptiveExpertSystem_39067022524924
// MI455X (gfx1250) — matrix operand order repaired and checked by reading, not compiled, not run on hardware
//
#include <hip/hip_runtime.h>
#include <stdint.h>

// ---------------------------------------------------------------------------
// MoE layer for MI455X (gfx1250): bf16 WMMA + Tensor Data Mover staging.
// Tokens = B*L = 4096, D = 1024, I = 4096, E = 8, top-2 routing.
// ---------------------------------------------------------------------------

typedef __attribute__((ext_vector_type(16))) __bf16 v16bf;
typedef __attribute__((ext_vector_type(8)))  float  v8f;
typedef unsigned int v4u __attribute__((ext_vector_type(4)));
typedef int          v8i __attribute__((ext_vector_type(8)));
typedef int          v4i __attribute__((ext_vector_type(4)));

#define D_DIM   1024
#define I_DIM   4096
#define E_NUM   8
#define NTOK    4096
#define LN_EPS  1e-5f

#define BM  128
#define BN  128
#define BK  32
#define LDT 40   // LDS row stride in bf16: 80 B = 64 B tile row + 16 B pad (16B aligned)

#if __has_builtin(__builtin_amdgcn_tensor_load_to_lds)
#define USE_TDM 1
#else
#define USE_TDM 0
#endif

union BFrag { v16bf v; uint4 q[2]; };

// A fragment (16x32 bf16, M x K), ISA 7.12.2 layout.
__device__ inline v16bf load_frag_a(const __bf16* base, int lane) {
    const int m  = lane & 15;
    const int kb = (lane >> 4) << 3;       // 0 or 8
    const __bf16* p = base + m * LDT + kb;
    BFrag f;
    f.q[0] = *(const uint4*)(p);           // K kb .. kb+7
    f.q[1] = *(const uint4*)(p + 16);      // K kb+16 .. kb+23
    return f.v;
}

// B fragment (32x16, K x N) from LDS holding the tile TRANSPOSED as [n][k].
__device__ inline v16bf load_frag_b(const __bf16* base, int lane) {
    const int n  = lane & 15;
    const int kb = (lane >> 4) << 3;       // 0 or 8
    const __bf16* p = base + n * LDT + kb;
    BFrag f;
    f.q[0] = *(const uint4*)(p);
    f.q[1] = *(const uint4*)(p + 16);
    return f.v;
}

#if USE_TDM
// TDM: DMA a [rows=128] x [k=32 bf16] tile (row stride = K elems) into LDS,
// inserting 16B of padding after every 64B row -> LDS row stride LDT=40 bf16.
__device__ inline void tdm_load_tile(uint32_t lds_off, const __bf16* gsrc, int K) {
    const uint64_t ga = (uint64_t)(uintptr_t)gsrc;
    v4u g0;
    g0[0] = 1u;                                        // count=1, user mode
    g0[1] = lds_off;                                   // lds_addr (bytes)
    g0[2] = (uint32_t)ga;                              // global_addr[31:0]
    g0[3] = ((uint32_t)(ga >> 32) & 0x01FFFFFFu)       // global_addr[56:32]
            | 0x80000000u;                             // type = 2 ("image")
    v8i g1;
    // data_size=1 (2B), pad_enable, pad_interval=3 (16 DW = 64B), pad_amount=3 (4 DW = 16B)
    g1[0] = (1 << 16) | (1 << 20) | (3 << 22) | (3 << 25);
    const uint32_t td0 = (uint32_t)K;                  // tensor_dim0 (no OOB clipping)
    const uint32_t td1 = 1u << 20;                     // tensor_dim1 large
    g1[1] = (int)((td0 & 0xFFFFu) << 16);              // [31:16] = tensor_dim0 lo
    g1[2] = (int)((td0 >> 16) | ((td1 & 0xFFFFu) << 16));
    g1[3] = (int)((td1 >> 16) | ((uint32_t)BK << 16)); // tile_dim0 = 32 (contiguous k)
    g1[4] = (int)BM;                                   // tile_dim1 = 128 rows, tile_dim2 = 0
    g1[5] = (int)K;                                    // tensor_dim0_stride[31:0]
    g1[6] = 0;
    g1[7] = 0;
    const v4i z4 = {0, 0, 0, 0};
    const v8i z8 = {0, 0, 0, 0, 0, 0, 0, 0};
    // clang-23 / therock toolchain: 6-arg form (g0, g1, g2, g3, extra, cpol)
    __builtin_amdgcn_tensor_load_to_lds(g0, g1, z4, z4, z8, 0);
}
#endif

// Synchronous fallback staging: 128x32 bf16 tile, 256 threads x 32 B.
__device__ inline void stage_tile(const __bf16* __restrict__ src, int row_stride,
                                  __bf16* __restrict__ dst, int tid) {
    const int row  = tid >> 1;
    const int half = tid & 1;
    const uint4* s = (const uint4*)(src + (size_t)row * row_stride + half * 16);
    uint4* d = (uint4*)(dst + row * LDT + half * 16);
    d[0] = s[0];
    d[1] = s[1];
}

__device__ inline float block_reduce(float v, float* sbuf, int tid) {
    sbuf[tid] = v;
    __syncthreads();
    for (int s = 128; s > 0; s >>= 1) {
        if (tid < s) sbuf[tid] += sbuf[tid + s];
        __syncthreads();
    }
    float r = sbuf[0];
    __syncthreads();
    return r;
}

// ---------------------------------------------------------------------------
// Router: shared LN stats, xhat (bf16), router logits, top-2 softmax weights.
// ---------------------------------------------------------------------------
__global__ __launch_bounds__(256)
void router_kernel(const float* __restrict__ x,
                   const float* __restrict__ rn_g, const float* __restrict__ rn_b,
                   const float* __restrict__ rw,   const float* __restrict__ rb,
                   float* __restrict__ combine, __bf16* __restrict__ xhat) {
    __shared__ float sbuf[256];
    const int tid = threadIdx.x;
    const int tok = blockIdx.x;
    const float* xp = x + (size_t)tok * D_DIM;

    float xs[4];
    float s = 0.f;
    #pragma unroll
    for (int j = 0; j < 4; ++j) { xs[j] = xp[tid + j * 256]; s += xs[j]; }
    const float mean = block_reduce(s, sbuf, tid) * (1.0f / D_DIM);

    float vs = 0.f;
    #pragma unroll
    for (int j = 0; j < 4; ++j) { float d = xs[j] - mean; vs += d * d; }
    const float var = block_reduce(vs, sbuf, tid) * (1.0f / D_DIM);
    const float rs = rsqrtf(var + LN_EPS);

    float p[E_NUM];
    #pragma unroll
    for (int e = 0; e < E_NUM; ++e) p[e] = 0.f;
    #pragma unroll
    for (int j = 0; j < 4; ++j) {
        const int i = tid + j * 256;
        const float xh = (xs[j] - mean) * rs;
        xhat[(size_t)tok * D_DIM + i] = (__bf16)xh;
        const float nx = xh * rn_g[i] + rn_b[i];
        #pragma unroll
        for (int e = 0; e < E_NUM; ++e) p[e] += nx * rw[e * D_DIM + i];
    }
    float logit[E_NUM];
    for (int e = 0; e < E_NUM; ++e) logit[e] = block_reduce(p[e], sbuf, tid) + rb[e];

    if (tid == 0) {
        int i1 = 0;
        for (int e = 1; e < E_NUM; ++e) if (logit[e] > logit[i1]) i1 = e;
        int i2 = (i1 == 0) ? 1 : 0;
        for (int e = 0; e < E_NUM; ++e)
            if (e != i1 && logit[e] > logit[i2]) i2 = e;
        const float eb = expf(logit[i2] - logit[i1]);   // <= 1
        const float wA = 1.0f / (1.0f + eb);
        const float wB = eb / (1.0f + eb);
        float* c = combine + (size_t)tok * E_NUM;
        #pragma unroll
        for (int e = 0; e < E_NUM; ++e) c[e] = 0.f;
        c[i1] = wA;
        c[i2] = wB;
    }
}

// Apply expert LN affine to shared xhat -> bf16 A operand for GEMM1.
__global__ __launch_bounds__(256)
void prep_a_kernel(const __bf16* __restrict__ xhat,
                   const float* __restrict__ g, const float* __restrict__ b,
                   __bf16* __restrict__ ae) {
    const int total = NTOK * D_DIM;
    for (int idx = blockIdx.x * 256 + threadIdx.x; idx < total; idx += gridDim.x * 256) {
        const int k = idx & (D_DIM - 1);
        ae[idx] = (__bf16)((float)xhat[idx] * g[k] + b[k]);
    }
}

// Transpose + fp32->bf16 convert: in[R x C] fp32  ->  out[C x R] bf16.
__global__ __launch_bounds__(256)
void transpose_cvt_kernel(const float* __restrict__ in, __bf16* __restrict__ outp,
                          int R, int C) {
    __shared__ float t[32][33];
    const int c0 = blockIdx.x * 32;
    const int r0 = blockIdx.y * 32;
    const int tx = threadIdx.x & 31;
    const int ty = threadIdx.x >> 5;   // 0..7
    #pragma unroll
    for (int i = 0; i < 32; i += 8)
        t[ty + i][tx] = in[(size_t)(r0 + ty + i) * C + c0 + tx];
    __syncthreads();
    #pragma unroll
    for (int i = 0; i < 32; i += 8)
        outp[(size_t)(c0 + ty + i) * R + r0 + tx] = (__bf16)t[tx][ty + i];
}

__global__ __launch_bounds__(256)
void zero_kernel(float* __restrict__ out, int n) {
    for (int i = blockIdx.x * 256 + threadIdx.x; i < n; i += gridDim.x * 256)
        out[i] = 0.0f;
}

// ---------------------------------------------------------------------------
// GEMM core: C = A[M x K] * Bt[N x K]^T, bf16 operands, fp32 WMMA accumulate.
// MODE 0: H1 = gelu(C + bias)             (bf16 store, output stride ldo = I)
// MODE 1: out += combine * (C + bias)     (fp32 RMW,   output stride ldo = D)
// Tile skip if all 128 token rows have combine == 0 (tiles identical in both
// GEMMs, so skipped/stale H1 rows are only ever multiplied by 0).
// ---------------------------------------------------------------------------
template <int MODE>
__global__ __launch_bounds__(256)
void gemm_kernel(const __bf16* __restrict__ A,    // [NTOK x K], row-major bf16
                 const __bf16* __restrict__ Bt,   // [N x K],    row-major bf16
                 const float*  __restrict__ bias, // [N]
                 const float*  __restrict__ cmb,  // combine + e (stride E_NUM)
                 __bf16* __restrict__ H1out,
                 float*  __restrict__ Fout,
                 int K, int ldo) {
    __shared__ __bf16 sA[2][BM * LDT];
    __shared__ __bf16 sB[2][BN * LDT];
    __shared__ int s_any;

    const int tid = threadIdx.x;
    const int m0 = blockIdx.y * BM;
    const int n0 = blockIdx.x * BN;

    if (tid == 0) s_any = 0;
    __syncthreads();
    if (tid < BM && cmb[(size_t)(m0 + tid) * E_NUM] != 0.0f) s_any = 1;
    __syncthreads();
    if (!s_any) return;                    // uniform: whole tile unrouted

    const int lane = tid & 31;
    const int wave = tid >> 5;
    const int wm = wave >> 2;              // 0..1 (64 rows)
    const int wn = wave & 3;               // 0..3 (32 cols)

    v8f acc[4][2] = {};
    const int nk = K / BK;

#if USE_TDM
    const bool w0 = (tid < 32);            // wave 0 drives the TDM
    int buf = 0;
    if (w0) {
        tdm_load_tile((uint32_t)(uintptr_t)&sA[0][0], A  + (size_t)m0 * K, K);
        tdm_load_tile((uint32_t)(uintptr_t)&sB[0][0], Bt + (size_t)n0 * K, K);
    }
    for (int ki = 0; ki < nk; ++ki) {
        if (w0) {
            if (ki + 1 < nk) {             // prefetch next tiles into buf^1
                const int kt = (ki + 1) * BK;
                tdm_load_tile((uint32_t)(uintptr_t)&sA[buf ^ 1][0], A  + (size_t)m0 * K + kt, K);
                tdm_load_tile((uint32_t)(uintptr_t)&sB[buf ^ 1][0], Bt + (size_t)n0 * K + kt, K);
                __builtin_amdgcn_s_wait_tensorcnt((short)2);  // current buf done (in-order)
            } else {
                __builtin_amdgcn_s_wait_tensorcnt((short)0);
            }
        }
        __syncthreads();

        v16bf af[4], bfm[2];
        #pragma unroll
        for (int i = 0; i < 4; ++i)
            af[i] = load_frag_a(&sA[buf][(wm * 64 + i * 16) * LDT], lane);
        #pragma unroll
        for (int j = 0; j < 2; ++j)
            bfm[j] = load_frag_b(&sB[buf][(wn * 32 + j * 16) * LDT], lane);
        #pragma unroll
        for (int i = 0; i < 4; ++i)
            #pragma unroll
            for (int j = 0; j < 2; ++j)
                acc[i][j] = __builtin_amdgcn_wmma_f32_16x16x32_bf16(
                    false, af[i], false, bfm[j], (short)0, acc[i][j], false, false);

        __syncthreads();                   // readers done before buf is re-filled
        buf ^= 1;
    }
#else
    for (int ki = 0; ki < nk; ++ki) {
        const int kt = ki * BK;
        stage_tile(A  + (size_t)m0 * K + kt, K, &sA[0][0], tid);
        stage_tile(Bt + (size_t)n0 * K + kt, K, &sB[0][0], tid);
        __syncthreads();

        v16bf af[4], bfm[2];
        #pragma unroll
        for (int i = 0; i < 4; ++i)
            af[i] = load_frag_a(&sA[0][(wm * 64 + i * 16) * LDT], lane);
        #pragma unroll
        for (int j = 0; j < 2; ++j)
            bfm[j] = load_frag_b(&sB[0][(wn * 32 + j * 16) * LDT], lane);
        #pragma unroll
        for (int i = 0; i < 4; ++i)
            #pragma unroll
            for (int j = 0; j < 2; ++j)
                acc[i][j] = __builtin_amdgcn_wmma_f32_16x16x32_bf16(
                    false, af[i], false, bfm[j], (short)0, acc[i][j], false, false);
        __syncthreads();
    }
#endif

    const int half = lane >> 4;
    const int ncol = lane & 15;
    #pragma unroll
    for (int i = 0; i < 4; ++i)
        #pragma unroll
        for (int j = 0; j < 2; ++j) {
            const int nn = n0 + wn * 32 + j * 16 + ncol;
            const float bv = bias[nn];
            #pragma unroll
            for (int r = 0; r < 8; ++r) {
                const int mm = m0 + wm * 64 + i * 16 + half * 8 + r;
                const float v = acc[i][j][r] + bv;
                if (MODE == 0) {
                    const float g = 0.5f * v * (1.0f + erff(v * 0.70710678118654752f));
                    H1out[(size_t)mm * ldo + nn] = (__bf16)g;
                } else {
                    const float cw = cmb[(size_t)mm * E_NUM];
                    // experts run sequentially; tiles disjoint -> plain RMW
                    Fout[(size_t)mm * ldo + nn] += cw * v;
                }
            }
        }
}

// ---------------------------------------------------------------------------
extern "C" void kernel_launch(void* const* d_in, const int* in_sizes, int n_in,
                              void* d_out, int out_size, void* d_ws, size_t ws_size,
                              hipStream_t stream) {
    (void)in_sizes; (void)n_in; (void)ws_size;
    const float* hidden   = (const float*)d_in[0];
    const float* rn_g     = (const float*)d_in[1];
    const float* rn_b     = (const float*)d_in[2];
    const float* router_w = (const float*)d_in[3];
    const float* router_b = (const float*)d_in[4];
    const float* ln_g     = (const float*)d_in[5];
    const float* ln_b     = (const float*)d_in[6];
    const float* w1       = (const float*)d_in[7];
    const float* b1       = (const float*)d_in[8];
    const float* w2       = (const float*)d_in[9];
    const float* b2       = (const float*)d_in[10];
    float* out = (float*)d_out;

    // Workspace layout (~64.13 MB total)
    char* ws = (char*)d_ws;
    __bf16* xhat    = (__bf16*)(ws);                             //  8 MB
    __bf16* ae      = (__bf16*)(ws + (size_t)8  * 1024 * 1024);  //  8 MB
    __bf16* h1      = (__bf16*)(ws + (size_t)16 * 1024 * 1024);  // 32 MB
    __bf16* w1t     = (__bf16*)(ws + (size_t)48 * 1024 * 1024);  //  8 MB  [I x D]
    __bf16* w2t     = (__bf16*)(ws + (size_t)56 * 1024 * 1024);  //  8 MB  [D x I]
    float*  combine = (float* )(ws + (size_t)64 * 1024 * 1024);  // 128 KB

    zero_kernel<<<1024, 256, 0, stream>>>(out, out_size);
    router_kernel<<<NTOK, 256, 0, stream>>>(hidden, rn_g, rn_b, router_w, router_b,
                                            combine, xhat);
    for (int e = 0; e < E_NUM; ++e) {
        prep_a_kernel<<<2048, 256, 0, stream>>>(
            xhat, ln_g + (size_t)e * D_DIM, ln_b + (size_t)e * D_DIM, ae);
        // w1[e]: [D x I] fp32 -> w1t: [I x D] bf16 ; w2[e]: [I x D] fp32 -> w2t: [D x I] bf16
        transpose_cvt_kernel<<<dim3(I_DIM / 32, D_DIM / 32), 256, 0, stream>>>(
            w1 + (size_t)e * D_DIM * I_DIM, w1t, D_DIM, I_DIM);
        transpose_cvt_kernel<<<dim3(D_DIM / 32, I_DIM / 32), 256, 0, stream>>>(
            w2 + (size_t)e * I_DIM * D_DIM, w2t, I_DIM, D_DIM);

        gemm_kernel<0><<<dim3(I_DIM / BN, NTOK / BM), 256, 0, stream>>>(
            ae, w1t, b1 + (size_t)e * I_DIM, combine + e, h1, nullptr, D_DIM, I_DIM);
        gemm_kernel<1><<<dim3(D_DIM / BN, NTOK / BM), 256, 0, stream>>>(
            h1, w2t, b2 + (size_t)e * D_DIM, combine + e, nullptr, out, I_DIM, D_DIM);
    }
}
